// GNN_1838246003277
// MI455X (gfx1250) — matrix operand order repaired and checked by reading, not compiled, not run on hardware
//
#include <hip/hip_runtime.h>

typedef __attribute__((ext_vector_type(16))) _Float16 v16h;
typedef __attribute__((ext_vector_type(2)))  _Float16 v2h;
typedef __attribute__((ext_vector_type(2)))  __fp16   v2hh;
typedef __attribute__((ext_vector_type(8)))  float    v8f;
typedef __attribute__((ext_vector_type(4)))  float    v4f;

#define NGRAPH 16

__global__ void __launch_bounds__(256)
egnn_zero_out(float* __restrict__ out, int n) {
  int i = blockIdx.x * blockDim.x + threadIdx.x;
  if (i < n) out[i] = 0.0f;
}

__global__ void __launch_bounds__(256)
egnn_layer_kernel(const float* __restrict__ edge_attr,
                  const int*   __restrict__ tgt,
                  const float* __restrict__ W1, const float* __restrict__ b1,
                  const float* __restrict__ W2, const float* __restrict__ b2,
                  const float* __restrict__ W3, const float* __restrict__ b3,
                  int E, unsigned long long mdiv, float* __restrict__ out)
{
  // per-wave h2 tile, f32, 16 rows x 64 cols padded to 68 (272 B row = 16-B aligned)
  __shared__ __align__(16) float sH2[8][16 * 68];
  __shared__ float sAB[8][16 * 2];                     // per-wave (a,b) patch
  __shared__ float sPool[NGRAPH * 3];                  // per-block graph pool

  const int tid  = threadIdx.x;
  const int lane = tid & 31;
  const int wave = tid >> 5;
  const int col  = lane & 15;   // N-column / edge row within tile
  const int kgrp = lane >> 4;   // K-half select per ISA lane layout

  if (tid < NGRAPH * 3) sPool[tid] = 0.0f;
  __syncthreads();

  // ---- loop-invariant register fragments (gathered from global; W2 slice
  // is 16KB and L2-resident, this runs once per block) --------------------
  // W2 as B-fragments: 4 N-tiles x 2 K-steps of 32x16 f16.
  // B layout per ISA 7.12.4: lane holds column (lane&15),
  // elements i=0..7 <-> K = 8*(lane>>4) + i, i=8..15 <-> K = 16 + 8*(lane>>4) + (i-8) within the 32-K step.
  v16h b2f[4][2];
#pragma unroll
  for (int t = 0; t < 4; ++t)
#pragma unroll
    for (int s = 0; s < 2; ++s)
#pragma unroll
      for (int i = 0; i < 16; ++i) {
        int k = 32 * s + ((i < 8) ? (8 * kgrp + i) : (16 + 8 * kgrp + (i - 8)));
        b2f[t][s][i] = (_Float16)W2[k * 64 + 16 * t + col];
      }

  // W3 (64x2) zero-padded to 64x16 B-fragments.
  v16h w3f[2];
#pragma unroll
  for (int s = 0; s < 2; ++s)
#pragma unroll
    for (int i = 0; i < 16; ++i) {
      int k = 32 * s + ((i < 8) ? (8 * kgrp + i) : (16 + 8 * kgrp + (i - 8)));
      w3f[s][i] = (col < 2) ? (_Float16)W3[k * 2 + col] : (_Float16)0.0f;
    }

  // W1/b1 in 16-bit A-fragment layout (ISA 7.12.2 16-bit A 16x32):
  // lanes 0-15: elements 0..7 -> K 0..7, 8..15 -> K 16..23; lanes 16-31: +8.
  v16h w1f[2], b1f[2];
#pragma unroll
  for (int s = 0; s < 2; ++s)
#pragma unroll
    for (int i = 0; i < 16; ++i) {
      int k = 32 * s + ((i < 8) ? (8 * kgrp + i) : (16 + 8 * kgrp + (i - 8)));
      w1f[s][i] = (_Float16)W1[k];
      b1f[s][i] = (_Float16)b1[k];
    }

  // Loop-invariant bias accumulators (read by WMMA SRC2, never copied).
  v8f cInit[4];
#pragma unroll
  for (int t = 0; t < 4; ++t) {
    float bi = b2[16 * t + col];
#pragma unroll
    for (int v = 0; v < 8; ++v) cInit[t][v] = bi;
  }
  v8f cabInit;
  {
    float b3i = (col < 2) ? b3[col] : 0.0f;
#pragma unroll
    for (int v = 0; v < 8; ++v) cabInit[v] = b3i;
  }

  const _Float16 zh = (_Float16)0.0f;

  // ---- grid-stride over edges, 16 edges per wave per round ------------
  const int waves  = (int)(gridDim.x * (blockDim.x >> 5));
  const int waveId = (int)(blockIdx.x * (blockDim.x >> 5)) + wave;
  const int stride = waves * 16;

  for (int base = waveId * 16; base < E; base += stride) {
    const int e = base + col;                // lanes L and L+16 mirror the same edge
    float dx = 0.f, dy = 0.f, dz = 0.f;
    int   tg = 0;
    if (e < E) {
      dx = edge_attr[3 * e + 0];
      dy = edge_attr[3 * e + 1];
      dz = edge_attr[3 * e + 2];
      tg = tgt[e];
    }
    {  // prefetch next round's edge data
      int en = e + stride;
      if (en < E) __builtin_prefetch(edge_attr + 3 * en, 0, 3);
    }
    const float r = sqrtf(dx * dx + dy * dy + dz * dz);
    const float scale = r / (r + 1e-8f);
    const _Float16 rh = (_Float16)r;

    // h1 = relu(r*W1 + b1), produced directly in A-fragment layout.
    v16h a0, a1;
#pragma unroll
    for (int i = 0; i < 16; ++i) {
      _Float16 v0 = rh * w1f[0][i] + b1f[0][i];
      _Float16 v1 = rh * w1f[1][i] + b1f[1][i];
      a0[i] = v0 > zh ? v0 : zh;
      a1[i] = v1 > zh ? v1 : zh;
    }

    // h1 @ W2 + b2: 8 x v_wmma_f32_16x16x32_f16, all 4 tiles live.
    v8f ct[4];
#pragma unroll
    for (int t = 0; t < 4; ++t)
      ct[t] = __builtin_amdgcn_wmma_f32_16x16x32_f16(false, a0, false, b2f[t][0],
                                                     (short)0, cInit[t], false, false);
#pragma unroll
    for (int t = 0; t < 4; ++t)
      ct[t] = __builtin_amdgcn_wmma_f32_16x16x32_f16(false, a1, false, b2f[t][1],
                                                     (short)0, ct[t], false, false);

    // Dump raw f32 C tiles to LDS (zero VALU; relu+cvt happen on reload).
    // C layout: VGPR v, lane -> row M=v+8*kgrp, col N=(lane&15)+16t.
    {
      float* rowq = &sH2[wave][(8 * kgrp) * 68 + col];
#pragma unroll
      for (int t = 0; t < 4; ++t)
#pragma unroll
        for (int v = 0; v < 8; ++v)
          rowq[v * 68 + 16 * t] = ct[t][v];
    }

    // Reload as f32 chunks (ds_load_b128 x8; same-wave DS ops are in-order),
    // then packed cvt directly on adjacent-K pairs + packed f16 relu.
    const float* rowp = &sH2[wave][col * 68];
    v4f q[8];
#pragma unroll
    for (int c = 0; c < 4; ++c) {       // K chunks at 0,16,32,48 (+8*kgrp)
      q[2 * c + 0] = *(const v4f*)(rowp + 16 * c + 8 * kgrp + 0);
      q[2 * c + 1] = *(const v4f*)(rowp + 16 * c + 8 * kgrp + 4);
    }
    v16h h0, h1;
#pragma unroll
    for (int c = 0; c < 4; ++c) {
#pragma unroll
      for (int d = 0; d < 2; ++d) {
#pragma unroll
        for (int u = 0; u < 2; ++u) {
          v2hh pr = __builtin_amdgcn_cvt_pkrtz(q[2 * c + d][2 * u + 0],
                                               q[2 * c + d][2 * u + 1]);
          v2h p = __builtin_bit_cast(v2h, pr);
          p[0] = p[0] > zh ? p[0] : zh;        // h2 relu (packed f16 max)
          p[1] = p[1] > zh ? p[1] : zh;
          int idx = 8 * (c & 1) + 4 * d + 2 * u;   // element within fragment
          if (c < 2) { h0[idx] = p[0]; h0[idx + 1] = p[1]; }
          else       { h1[idx] = p[0]; h1[idx + 1] = p[1]; }
        }
      }
    }

    // ab = h2 @ W3 + b3 (padded to 16 cols): 2 WMMAs.
    v8f cab;
    cab = __builtin_amdgcn_wmma_f32_16x16x32_f16(false, h0, false, w3f[0],
                                                 (short)0, cabInit, false, false);
    cab = __builtin_amdgcn_wmma_f32_16x16x32_f16(false, h1, false, w3f[1],
                                                 (short)0, cab, false, false);

    // Columns 0/1 of C hold (a,b): lanes 0,1,16,17 publish to LDS.
    if (col < 2) {
#pragma unroll
      for (int v = 0; v < 8; ++v)
        sAB[wave][(v + 8 * kgrp) * 2 + col] = cab[v];
    }

    // msg = (a + b*scale) * edge_attr, scattered straight into graph pool.
    if (lane < 16 && e < E) {
      const float av = sAB[wave][lane * 2 + 0];
      const float bv = sAB[wave][lane * 2 + 1];
      const float f  = av + bv * scale;
      int g = (int)(((unsigned long long)(unsigned int)tg * mdiv) >> 45);  // tg/npg
      g = g > (NGRAPH - 1) ? (NGRAPH - 1) : g;
      atomicAdd(&sPool[g * 3 + 0], f * dx);
      atomicAdd(&sPool[g * 3 + 1], f * dy);
      atomicAdd(&sPool[g * 3 + 2], f * dz);
    }
  }

  __syncthreads();
  if (tid < NGRAPH * 3) unsafeAtomicAdd(&out[tid], sPool[tid]);
}

extern "C" void kernel_launch(void* const* d_in, const int* in_sizes, int n_in,
                              void* d_out, int out_size, void* d_ws, size_t ws_size,
                              hipStream_t stream) {
  (void)n_in; (void)d_ws; (void)ws_size;
  const float* edge_attr  = (const float*)d_in[1];
  const float* W1         = (const float*)d_in[2];
  const float* b1         = (const float*)d_in[3];
  const float* W2         = (const float*)d_in[4];
  const float* b2         = (const float*)d_in[5];
  const float* W3         = (const float*)d_in[6];
  const float* b3         = (const float*)d_in[7];
  const int*   edge_index = (const int*)d_in[8];

  const int E = in_sizes[1] / 3;        // edge_attr [E,3]
  const int N = in_sizes[0] / 3;        // x [N,3]
  const int L = in_sizes[7] / 2;        // b3 [L,2]
  const int layer = L - 1;              // only the last layer is live
  const int npg = N / NGRAPH;
  // exact magic-divide for unsigned t/npg (t < 2^16, npg ~ 3125):
  const unsigned long long mdiv = ((1ULL << 45) + (unsigned)npg - 1) / (unsigned)npg;

  // Zero output (harness poisons it; we accumulate atomically).
  egnn_zero_out<<<(out_size + 255) / 256, 256, 0, stream>>>((float*)d_out, out_size);

  int wavesNeeded = (E + 15) / 16;
  int blocks = (wavesNeeded + 7) / 8;
  if (blocks > 640) blocks = 640;
  if (blocks < 1) blocks = 1;

  egnn_layer_kernel<<<blocks, 256, 0, stream>>>(
      edge_attr, edge_index + E,
      W1 + layer * 64,       b1 + layer * 64,
      W2 + layer * 64 * 64,  b2 + layer * 64,
      W3 + layer * 64 * 2,   b3 + layer * 2,
      E, mdiv, (float*)d_out);
}
